// LSTM_62148176773386
// MI455X (gfx1250) — matrix operand order repaired and checked by reading, not compiled, not run on hardware
//
#include <hip/hip_runtime.h>
#include <math.h>

#define T_STEPS 256
#define D_IN    56
#define HID     64
#define G4      256   // 4*HID
#define FC_D    32
#define NCLS    6

typedef __attribute__((ext_vector_type(16))) __bf16 v16bf;
typedef __attribute__((ext_vector_type(8)))  __bf16 v8bf;
typedef __attribute__((ext_vector_type(8)))  float  v8f;

// Pack two contiguous 8-element bf16 chunks into the 16-element WMMA operand.
// For both A and B 16-bit layouts, elements [0..7] are the "low" contiguous K
// chunk and [8..15] the "high" chunk (CDNA5 ISA 7.12.2).
static __device__ inline v16bf pack2(v8bf lo, v8bf hi) {
  v16bf r;
#pragma unroll
  for (int e = 0; e < 8; ++e) { r[e] = lo[e]; r[8 + e] = hi[e]; }
  return r;
}

// B fragment: lane holds column n, chunks K = kbase..kbase+7 and kbase+16..kbase+23 in LDS row.
static __device__ inline v16bf load_bfrag(const __bf16* W, int n, int kbase) {
  const v8bf* p = reinterpret_cast<const v8bf*>(W + n * HID + kbase);
  return pack2(p[0], p[2]);
}

// A fragment from LDS h tile: lane holds row m; chunks at kcol and kcol+16.
static __device__ inline v16bf load_afrag_lds(const __bf16* hrow, int kcol) {
  v8bf c0 = *reinterpret_cast<const v8bf*>(hrow + kcol);
  v8bf c1 = *reinterpret_cast<const v8bf*>(hrow + kcol + 16);
  return pack2(c0, c1);
}

// 8 contiguous f32 from global -> 8 bf16 (two aligned float4 loads).
static __device__ inline v8bf cvt8(const float* p) {
  float4 a = *reinterpret_cast<const float4*>(p);
  float4 b = *reinterpret_cast<const float4*>(p + 4);
  v8bf r;
  r[0] = (__bf16)a.x; r[1] = (__bf16)a.y; r[2] = (__bf16)a.z; r[3] = (__bf16)a.w;
  r[4] = (__bf16)b.x; r[5] = (__bf16)b.y; r[6] = (__bf16)b.z; r[7] = (__bf16)b.w;
  return r;
}

static __device__ inline float fast_sigmoid(float x) { return 1.0f / (1.0f + __expf(-x)); }
static __device__ inline float fast_tanh(float x)    { return 2.0f / (1.0f + __expf(-2.0f * x)) - 1.0f; }

#define WMMA_BF16(A, B, C) \
  __builtin_amdgcn_wmma_f32_16x16x32_bf16(false, (A), false, (B), (short)0, (C), false, false)

__global__ __launch_bounds__(128)
void lstm_wmma_fused(const float* __restrict__ x,
                     const float* __restrict__ W_ih, const float* __restrict__ W_hh,
                     const float* __restrict__ b_ih, const float* __restrict__ b_hh,
                     const float* __restrict__ fc_w, const float* __restrict__ fc_b,
                     const float* __restrict__ fc2_w, const float* __restrict__ fc2_b,
                     float* __restrict__ out, int Bsz)
{
  extern __shared__ char smem[];
  __bf16* sWih = (__bf16*)smem;                 // [256][64] bf16 (K padded 56->64)
  __bf16* sWhh = sWih + G4 * HID;               // [256][64] bf16
  __bf16* sFc  = sWhh + G4 * HID;               // [32][64]  bf16
  __bf16* sH   = sFc + FC_D * HID;              // [4 waves][16][64] bf16
  float*  sO1  = (float*)(sH + 4 * 16 * HID);   // [4 waves][16][32] f32

  const int tid  = threadIdx.x;
  const int wave = tid >> 5;
  const int lane = tid & 31;
  const int nl   = lane & 15;     // column / row-in-tile index
  const int hi   = lane >> 4;     // lane half
  const int kb8  = hi * 8;        // A-fragment per-lane K offset
  const int kb16 = hi * 8;        // B-fragment per-lane K offset

  // ---- Stage weights into LDS as bf16; zero initial h ----
  for (int i = tid; i < G4 * HID; i += 128) {
    int k = i & (HID - 1);
    sWih[i] = (k < D_IN) ? (__bf16)W_ih[(i >> 6) * D_IN + k] : (__bf16)0.0f;
    sWhh[i] = (__bf16)W_hh[i];
  }
  for (int i = tid; i < FC_D * HID; i += 128) sFc[i] = (__bf16)fc_w[i];
  for (int i = tid; i < 4 * 16 * HID; i += 128) sH[i] = (__bf16)0.0f;
  __syncthreads();

  // Per-lane fused bias for the 16 gate tiles (n = gt*16 + nl)
  float bias[16];
#pragma unroll
  for (int gt = 0; gt < 16; ++gt) {
    int n = gt * 16 + nl;
    bias[gt] = b_ih[n] + b_hh[n];
  }

  const int rowBase = blockIdx.x * 64 + wave * 16;        // batch rows of this wave
  const float* xrow = x + (size_t)(rowBase + nl) * T_STEPS * D_IN;
  __bf16* sHw = sH + wave * 16 * HID;                     // wave-private h tile
  const __bf16* hrow = sHw + nl * HID;                    // A-layout row (m = nl)

  v8f cc[4];                                              // cell state, D layout
#pragma unroll
  for (int j = 0; j < 4; ++j)
#pragma unroll
    for (int r = 0; r < 8; ++r) cc[j][r] = 0.0f;

#pragma unroll 1
  for (int t = 0; t < T_STEPS; ++t) {
    // h_{t-1} A-fragments (wave-private LDS; LDS ops from one wave are in-order)
    v16bf ah0 = load_afrag_lds(hrow, 0 + kb8);
    v16bf ah1 = load_afrag_lds(hrow, 32 + kb8);

    // x_t A-fragments straight from global, f32 -> bf16
    const float* xt = xrow + t * D_IN;
    v8bf x00 = cvt8(xt + kb8);             // K = kb8 .. kb8+7
    v8bf x01 = cvt8(xt + kb8 + 16);        // K = kb8+16 .. kb8+23
    v8bf x10 = cvt8(xt + 32 + kb8);        // K = 32+kb8 ..
    v8bf x11;
    if (hi == 0) {
      x11 = cvt8(xt + 48);                 // K = 48..55
    } else {
#pragma unroll
      for (int e = 0; e < 8; ++e) x11[e] = (__bf16)0.0f;  // K=56..63 zero pad
    }
    v16bf ax0 = pack2(x00, x01);
    v16bf ax1 = pack2(x10, x11);

    if (t + 1 < T_STEPS) __builtin_prefetch(xt + D_IN, 0, 1);  // global_prefetch_b8

#pragma unroll
    for (int j = 0; j < 4; ++j) {          // hidden tile (16 cols of H=64)
      v8f g[4];
      // Software-pipelined B-fragment preload: issue next gate's ds_loads
      // (distinct registers) before the current gate's chained WMMAs, so LDS
      // latency hides under the matrix pipe.
      int n0 = j * 16 + nl;                // gate 0 tile = j
      v16bf c_ih0 = load_bfrag(sWih, n0, 0  + kb16);
      v16bf c_ih1 = load_bfrag(sWih, n0, 32 + kb16);
      v16bf c_hh0 = load_bfrag(sWhh, n0, 0  + kb16);
      v16bf c_hh1 = load_bfrag(sWhh, n0, 32 + kb16);
#pragma unroll
      for (int gate = 0; gate < 4; ++gate) {   // i, f, g, o
        v16bf n_ih0, n_ih1, n_hh0, n_hh1;
        if (gate < 3) {
          int nn = ((gate + 1) * 4 + j) * 16 + nl;
          n_ih0 = load_bfrag(sWih, nn, 0  + kb16);
          n_ih1 = load_bfrag(sWih, nn, 32 + kb16);
          n_hh0 = load_bfrag(sWhh, nn, 0  + kb16);
          n_hh1 = load_bfrag(sWhh, nn, 32 + kb16);
        }
        v8f acc;
#pragma unroll
        for (int r = 0; r < 8; ++r) acc[r] = 0.0f;   // bias folded in later
        acc = WMMA_BF16(ax0, c_ih0, acc);
        acc = WMMA_BF16(ax1, c_ih1, acc);
        acc = WMMA_BF16(ah0, c_hh0, acc);
        acc = WMMA_BF16(ah1, c_hh1, acc);
        g[gate] = acc;
        if (gate < 3) { c_ih0 = n_ih0; c_ih1 = n_ih1; c_hh0 = n_hh0; c_hh1 = n_hh1; }
      }
      // Elementwise LSTM cell update in D layout (bias added here, not splatted
      // into accumulators); write h back (bf16) to LDS.
      const float bi = bias[0 * 4 + j], bf = bias[1 * 4 + j];
      const float bg = bias[2 * 4 + j], bo = bias[3 * 4 + j];
#pragma unroll
      for (int r = 0; r < 8; ++r) {
        float i_t = fast_sigmoid(g[0][r] + bi);
        float f_t = fast_sigmoid(g[1][r] + bf);
        float g_t = fast_tanh(g[2][r] + bg);
        float o_t = fast_sigmoid(g[3][r] + bo);
        float c_n = f_t * cc[j][r] + i_t * g_t;
        cc[j][r]  = c_n;
        float h_n = o_t * fast_tanh(c_n);
        sHw[(r + 8 * hi) * HID + j * 16 + nl] = (__bf16)h_n;
      }
    }
    // Compiler-only barrier: keep h stores ordered before next iteration's
    // loads (hardware LDS is in-order per wave; no s_wait drain needed).
    asm volatile("" ::: "memory");
  }

  // ---- FC1: relu(h_last @ fc_w^T + fc_b) via WMMA (N=32 -> 2 tiles) ----
  v16bf ah0 = load_afrag_lds(hrow, 0 + kb8);
  v16bf ah1 = load_afrag_lds(hrow, 32 + kb8);
  float* sO1w = sO1 + wave * 16 * FC_D;
#pragma unroll
  for (int nt = 0; nt < 2; ++nt) {
    float fb = fc_b[nt * 16 + nl];
    v16bf bw0 = load_bfrag(sFc, nt * 16 + nl, 0  + kb16);
    v16bf bw1 = load_bfrag(sFc, nt * 16 + nl, 32 + kb16);
    v8f acc;
#pragma unroll
    for (int r = 0; r < 8; ++r) acc[r] = fb;
    acc = WMMA_BF16(ah0, bw0, acc);
    acc = WMMA_BF16(ah1, bw1, acc);
#pragma unroll
    for (int r = 0; r < 8; ++r)
      sO1w[(r + 8 * hi) * FC_D + nt * 16 + nl] = fmaxf(acc[r], 0.0f);
  }
  asm volatile("" ::: "memory");

  // ---- FC2: out[16][6] = relu_out @ fc2_w^T + fc2_b (tiny, per-lane dots) ----
  for (int task = lane; task < 16 * NCLS; task += 32) {
    int row = task / NCLS;
    int cls = task - row * NCLS;
    float sum = fc2_b[cls];
#pragma unroll
    for (int k = 0; k < FC_D; ++k)
      sum += sO1w[row * FC_D + k] * fc2_w[cls * FC_D + k];
    out[(size_t)(rowBase + row) * NCLS + cls] = sum;
  }
}

extern "C" void kernel_launch(void* const* d_in, const int* in_sizes, int n_in,
                              void* d_out, int out_size, void* d_ws, size_t ws_size,
                              hipStream_t stream) {
  const float* x     = (const float*)d_in[0];
  const float* W_ih  = (const float*)d_in[1];
  const float* W_hh  = (const float*)d_in[2];
  const float* b_ih  = (const float*)d_in[3];
  const float* b_hh  = (const float*)d_in[4];
  const float* fc_w  = (const float*)d_in[5];
  const float* fc_b  = (const float*)d_in[6];
  const float* fc2_w = (const float*)d_in[7];
  const float* fc2_b = (const float*)d_in[8];
  float* out = (float*)d_out;

  int Bsz = in_sizes[0] / (T_STEPS * D_IN);     // 4096
  int grid = Bsz / 64;                          // 64 batch rows per block (4 waves x 16)

  size_t shmem = (size_t)(G4 * HID        /* sWih */
                        + G4 * HID        /* sWhh */
                        + FC_D * HID      /* sFc  */
                        + 4 * 16 * HID    /* sH   */) * sizeof(__bf16)
               + (size_t)(4 * 16 * FC_D) * sizeof(float);  /* sO1 */

  lstm_wmma_fused<<<grid, 128, shmem, stream>>>(x, W_ih, W_hh, b_ih, b_hh,
                                                fc_w, fc_b, fc2_w, fc2_b, out, Bsz);
}
